// GSMLayer_27633819582611
// MI455X (gfx1250) — matrix operand order repaired and checked by reading, not compiled, not run on hardware
//
#include <hip/hip_runtime.h>

// CDNA5 / gfx1250 WMMA types
typedef __attribute__((ext_vector_type(16))) __bf16 v16bf;
typedef __attribute__((ext_vector_type(8)))  float  v8f;

union BF16x16 { unsigned int u[8]; uint4 q[2]; v16bf v; };

// pack two f32 -> packed bf16 pair in ONE v_perm_b32 (truncating)
__device__ __forceinline__ unsigned int pack_bf16(float lo, float hi) {
    return __builtin_amdgcn_perm(__builtin_bit_cast(unsigned int, hi),
                                 __builtin_bit_cast(unsigned int, lo),
                                 0x07060302u);
}
// bf16 halves -> f32: pure bit ops
__device__ __forceinline__ float bf_lo(unsigned int u) { return __builtin_bit_cast(float, u << 16); }
__device__ __forceinline__ float bf_hi(unsigned int u) { return __builtin_bit_cast(float, u & 0xFFFF0000u); }

// fast softplus + sigmoid on native trans pipes (v_exp_f32/v_log_f32/v_rcp_f32)
#define LOG2E_F 1.4426950408889634f
#define LN2_F   0.6931471805599453f
__device__ __forceinline__ void softsig(float z, float& h, float& s) {
    float zc = fminf(z, 20.f);
    float t  = __builtin_amdgcn_exp2f(zc * LOG2E_F);
    float u  = 1.f + t;
    s = t * __builtin_amdgcn_rcpf(u);
    float l  = __builtin_amdgcn_logf(u) * LN2_F;
    h = (z > 20.f) ? z : l;
}
__device__ __forceinline__ float sigf(float z) {
    float zc = fminf(z, 20.f);
    float t  = __builtin_amdgcn_exp2f(zc * LOG2E_F);
    return t * __builtin_amdgcn_rcpf(1.f + t);
}

// One wave (32 lanes) owns a 16-row batch tile and scans T steps.
// Lane l: row = l&15, hf = l>>4.
// A-operand (16x32 bf16): chunk c, vgpr v holds cols (c*32 + k0, +1),
//   k0 = (v<4 ? 2v : 16+2(v-4)) + 8*hf.
// B-operand (32x16 bf16): N = l&15, vgpr v holds K = k0, +1 (k0 as for A).
// D (16x16 f32): vgpr g holds M = g+8*hf, N = l&15 (+16 per N-tile).
__launch_bounds__(32, 1)
__global__ void gsm_scan_wmma_kernel(const float* __restrict__ inputs,  // (B,T,2)
                                     const float* __restrict__ gamma0,  // (B,1)
                                     const float* __restrict__ W1,      // (2,64)
                                     const float* __restrict__ b1,      // (64)
                                     const float* __restrict__ W2,      // (64,64)
                                     const float* __restrict__ b2,      // (64)
                                     const float* __restrict__ W3,      // (64,1)
                                     float* __restrict__ out,           // (B,T)
                                     float* __restrict__ gamma_last,    // (B)
                                     int T) {
    // LDS: activation transpose buffer + per-lane B-operand streams for W2 / W2^T.
    // Stream layout: [lane][64 dwords], row stride 68 dwords (16B-aligned, bank spread).
    __shared__ float        xpose[16 * 66];
    __shared__ unsigned int w2s [32 * 68];
    __shared__ unsigned int w2ts[32 * 68];

    const int lane = (int)threadIdx.x;     // 0..31
    const int row  = lane & 15;
    const int hf   = lane >> 4;
    const int grow = (int)blockIdx.x * 16 + row;   // global batch row

    // ---------- one-time per-lane constant staging ----------
    // Layer-1: (W1[0,col], W1[1,col]) bf16-packed per column; b1 pairs packed.
    unsigned int w1ab[32], b1p[16];
    #pragma unroll
    for (int c = 0; c < 2; ++c) {
        #pragma unroll
        for (int v = 0; v < 8; ++v) {
            int k0  = (v < 4) ? (2 * v) : (16 + 2 * (v - 4));
            int col = c * 32 + k0 + 8 * hf;
            int i   = c * 8 + v;
            w1ab[2*i]   = pack_bf16(W1[col],     W1[64 + col]);
            w1ab[2*i+1] = pack_bf16(W1[col + 1], W1[64 + col + 1]);
            b1p[i]      = pack_bf16(b1[col],     b1[col + 1]);
        }
    }

    // (W3[n], b2[n]) packed per N-tile (D-layout columns)
    unsigned int w3b2[4];
    #pragma unroll
    for (int nt = 0; nt < 4; ++nt) {
        int n = nt * 16 + row;
        w3b2[nt] = pack_bf16(W3[n], b2[n]);
    }

    // Stage W2 (z2 = h1@W2) and W2^T (dh1 = d2@W2^T) bf16 B-operand streams in LDS.
    // Block blk = kc*4 + nt occupies dwords [blk*8, blk*8+8) of this lane's row.
    #pragma unroll
    for (int kc = 0; kc < 2; ++kc) {
        #pragma unroll
        for (int nt = 0; nt < 4; ++nt) {
            int n = nt * 16 + row;
            #pragma unroll
            for (int v = 0; v < 8; ++v) {
                int k0 = kc * 32 + ((v < 4) ? (2 * v) : (16 + 2 * (v - 4))) + hf * 8;
                int o  = lane * 68 + (kc * 4 + nt) * 8 + v;
                w2s [o] = pack_bf16(W2[k0 * 64 + n], W2[(k0 + 1) * 64 + n]);
                w2ts[o] = pack_bf16(W2[n * 64 + k0], W2[n * 64 + k0 + 1]);
            }
        }
    }
    __syncthreads();

    float gamma = gamma0[grow];
    const float2* __restrict__ inp = (const float2*)(inputs + (size_t)grow * T * 2);
    float* __restrict__ outp = out + (size_t)grow * T;
    const float G = 1.0f;

    // ---------- sequential scan over T ----------
    for (int t = 0; t < T; ++t) {
        float2 ed  = inp[t];           // (eps, dt)
        float  eps = ed.x, dtv = ed.y;

        // Layer 1: z1 = [eps,gamma]@W1 + b1 ; h1 = softplus (-> bf16 A), s1 = sigmoid.
        BF16x16 a1[2];
        unsigned int s1p[16];          // sigmoid(z1) for backward, bf16-packed
        #pragma unroll
        for (int i = 0; i < 16; ++i) {
            unsigned int wa = w1ab[2*i], wb = w1ab[2*i+1], bb = b1p[i];
            float z0 = fmaf(gamma, bf_hi(wa), fmaf(eps, bf_lo(wa), bf_lo(bb)));
            float z1 = fmaf(gamma, bf_hi(wb), fmaf(eps, bf_lo(wb), bf_hi(bb)));
            float h0, s0, h1, s1;
            softsig(z0, h0, s0);
            softsig(z1, h1, s1);
            s1p[i] = pack_bf16(s0, s1);
            a1[i >> 3].u[i & 7] = pack_bf16(h0, h1);
        }

        // GEMM1: z2 = h1 @ W2 (8 WMMAs); B operands streamed from LDS per tile.
        v8f acc[4];
        #pragma unroll
        for (int nt = 0; nt < 4; ++nt) {
            BF16x16 bk0, bk1;
            bk0.q[0] = *(const uint4*)&w2s[lane * 68 + nt * 8];
            bk0.q[1] = *(const uint4*)&w2s[lane * 68 + nt * 8 + 4];
            bk1.q[0] = *(const uint4*)&w2s[lane * 68 + (4 + nt) * 8];
            bk1.q[1] = *(const uint4*)&w2s[lane * 68 + (4 + nt) * 8 + 4];
            v8f z = {};
            z = __builtin_amdgcn_wmma_f32_16x16x32_bf16(false, a1[0].v, false, bk0.v,
                                                        (short)0, z, false, false);
            z = __builtin_amdgcn_wmma_f32_16x16x32_bf16(false, a1[1].v, false, bk1.v,
                                                        (short)0, z, false, false);
            acc[nt] = z;
        }

        // d2 = W3 * sigmoid(z2 + b2); stage to LDS (m,n) for layout transpose
        #pragma unroll
        for (int nt = 0; nt < 4; ++nt) {
            float w3v = bf_lo(w3b2[nt]);
            float b2v = bf_hi(w3b2[nt]);
            #pragma unroll
            for (int g = 0; g < 8; ++g) {
                float s2 = sigf(acc[nt][g] + b2v);
                xpose[(g + 8 * hf) * 66 + (row + 16 * nt)] = w3v * s2;
            }
        }
        __syncthreads();   // single-wave WG: cheap DS ordering

        // Re-read d2 in A-operand (row) layout as float2, pack to bf16
        BF16x16 a2[2];
        #pragma unroll
        for (int c = 0; c < 2; ++c) {
            #pragma unroll
            for (int v = 0; v < 8; ++v) {
                int k0  = (v < 4) ? (2 * v) : (16 + 2 * (v - 4));
                int col = c * 32 + k0 + 8 * hf;
                float2 dd = *(const float2*)&xpose[row * 66 + col];
                a2[c].u[v] = pack_bf16(dd.x, dd.y);
            }
        }
        __syncthreads();

        // GEMM2: dh1 = d2 @ W2^T (8 WMMAs); B from LDS.
        v8f dacc[4];
        #pragma unroll
        for (int nt = 0; nt < 4; ++nt) {
            BF16x16 bk0, bk1;
            bk0.q[0] = *(const uint4*)&w2ts[lane * 68 + nt * 8];
            bk0.q[1] = *(const uint4*)&w2ts[lane * 68 + nt * 8 + 4];
            bk1.q[0] = *(const uint4*)&w2ts[lane * 68 + (4 + nt) * 8];
            bk1.q[1] = *(const uint4*)&w2ts[lane * 68 + (4 + nt) * 8 + 4];
            v8f z = {};
            z = __builtin_amdgcn_wmma_f32_16x16x32_bf16(false, a2[0].v, false, bk0.v,
                                                        (short)0, z, false, false);
            z = __builtin_amdgcn_wmma_f32_16x16x32_bf16(false, a2[1].v, false, bk1.v,
                                                        (short)0, z, false, false);
            dacc[nt] = z;
        }

        // Transpose dh1 (D layout -> row layout) via LDS
        #pragma unroll
        for (int nt = 0; nt < 4; ++nt) {
            #pragma unroll
            for (int g = 0; g < 8; ++g) {
                xpose[(g + 8 * hf) * 66 + (row + 16 * nt)] = dacc[nt][g];
            }
        }
        __syncthreads();

        // jac = (dh1 * s1) @ W1^T — per-lane partial over its 32 columns
        float je = 0.f, jg = 0.f;
        #pragma unroll
        for (int c = 0; c < 2; ++c) {
            #pragma unroll
            for (int v = 0; v < 8; ++v) {
                int k0  = (v < 4) ? (2 * v) : (16 + 2 * (v - 4));
                int col = c * 32 + k0 + 8 * hf;
                int i   = c * 8 + v;
                unsigned int wa = w1ab[2*i], wb = w1ab[2*i+1], sp = s1p[i];
                float2 dd = *(const float2*)&xpose[row * 66 + col];
                float d0 = dd.x * bf_lo(sp);
                float d1 = dd.y * bf_hi(sp);
                je = fmaf(d0, bf_lo(wa), je);  je = fmaf(d1, bf_lo(wb), je);
                jg = fmaf(d0, bf_hi(wa), jg);  jg = fmaf(d1, bf_hi(wb), jg);
            }
        }
        __syncthreads();

        // merge the two half-wave partials; both halves get the full sums
        je += __shfl_xor(je, 16, 32);
        jg += __shfl_xor(jg, 16, 32);

        if (hf == 0) outp[t] = jg;            // output = dE/dgamma
        gamma = fmaf(dtv, -G * je, gamma);    // recurrence uses dE/deps
    }

    if (hf == 0) gamma_last[grow] = gamma;
}

extern "C" void kernel_launch(void* const* d_in, const int* in_sizes, int n_in,
                              void* d_out, int out_size, void* d_ws, size_t ws_size,
                              hipStream_t stream) {
    const float* inputs = (const float*)d_in[0];
    const float* gamma0 = (const float*)d_in[1];
    const float* W1     = (const float*)d_in[2];
    const float* b1     = (const float*)d_in[3];
    const float* W2     = (const float*)d_in[4];
    const float* b2     = (const float*)d_in[5];
    const float* W3     = (const float*)d_in[6];
    // d_in[7] = b3: constant offset, vanishes in the gradient — unused.

    const int B = in_sizes[1];                // gamma0 flat count == B
    const int T = in_sizes[0] / (2 * B);      // inputs = B*T*2

    float* out        = (float*)d_out;              // [B*T] jac_gamma
    float* gamma_last = out + (size_t)B * T;        // [B]

    dim3 grid(B / 16), block(32);
    gsm_scan_wmma_kernel<<<grid, block, 0, stream>>>(inputs, gamma0, W1, b1, W2, b2, W3,
                                                     out, gamma_last, T);
}
